// RCGNN_18279380812412
// MI455X (gfx1250) — matrix operand order repaired and checked by reading, not compiled, not run on hardware
//
#include <hip/hip_runtime.h>

// ---------------------------------------------------------------------------
// R-GCN forward for MI455X (gfx1250, wave32, WMMA + async-LDS DMA).
//
// Sizes (fixed by the reference):
#define NN     50000
#define EE     800000
#define FN     16
#define RR     4
#define HH     128
#define DEPTH  4
#define GG     64
//
// Math: per layer  h' = sum_r (mean_{src->n, rel r} h[src]) @ W_r
//                       + h @ W_root + b        (ReLU except last layer)
// The linear transform commutes past the mean, so we aggregate raw h per
// (dst, rel) segment with f32 HW atomics, then do 5 bf16-WMMA GEMMs fused
// with 1/cnt scaling + bias + ReLU.  h is stored bf16 (the GEMM rounds to
// bf16 anyway) which halves the dominant per-edge gather traffic.
// ---------------------------------------------------------------------------

typedef __attribute__((ext_vector_type(16))) __bf16        v16bf;
typedef __attribute__((ext_vector_type(4)))  __bf16        v4bf;
typedef __attribute__((ext_vector_type(8)))  float         v8f;
typedef __attribute__((ext_vector_type(4)))  float         v4f;
typedef __attribute__((ext_vector_type(4)))  unsigned int  v4u;

// ---- WMMA fragment helpers (layouts per CDNA5 ISA 7.12.2, wave32) ----------

// A fragment, 16x32 bf16, from an LDS tile with row stride `strideBf` (bf16
// elements).  lanes 0-15: M=lane, K = {kb..kb+7, kb+16..kb+23};
// lanes 16-31: M=lane-16, K = {kb+8..kb+15, kb+24..kb+31}.
__device__ __forceinline__ v16bf lds_a_frag(const __bf16* base, int strideBf, int kt) {
    int lane  = threadIdx.x & 31;
    int m     = lane & 15;
    int khalf = (lane >> 4) ? 8 : 0;
    const __bf16* p = base + m * strideBf + kt * 32 + khalf;
    union { v4u u[2]; v16bf v; } un;
    un.u[0] = *(const v4u*)(p);        // K chunk +0..7
    un.u[1] = *(const v4u*)(p + 16);   // K chunk +16..23
    return un.v;
}

// B fragment, 32x16 bf16, from global weights pre-packed TRANSPOSED:
// Wt[n][k] (row stride Kp bf16).  lanes 0-15: N=ncol0+lane, K = kt*32+{0..7,16..23};
// lanes 16-31: N=ncol0+lane-16, K = kt*32+{8..15,24..31}.
__device__ __forceinline__ v16bf gbl_b_frag(const __bf16* Wt, int Kp, int ncol0, int kt) {
    int lane = threadIdx.x & 31;
    int n    = ncol0 + (lane & 15);
    int kb   = kt * 32 + ((lane >> 4) ? 8 : 0);
    const __bf16* p = Wt + (size_t)n * Kp + kb;
    union { v4u u[2]; v16bf v; } un;
    un.u[0] = *(const v4u*)(p);
    un.u[1] = *(const v4u*)(p + 16);
    return un.v;
}

// ---- utility kernels -------------------------------------------------------

__global__ void zero4_kernel(float* p, size_t n4) {   // n4 = count of float4
    size_t i = (size_t)blockIdx.x * blockDim.x + threadIdx.x;
    size_t s = (size_t)gridDim.x * blockDim.x;
    v4f z = {};
    for (; i < n4; i += s) ((v4f*)p)[i] = z;
}

// Pack fp32 weight W[mat][K][Ncols] (row-major over K) into transposed,
// zero-padded bf16 Wt[mat][Ncols][Kp].
__global__ void pack_bt_kernel(const float* __restrict__ W, __bf16* __restrict__ out,
                               int K, int Ncols, int Kp) {
    int mat = blockIdx.y;
    const float* Wm = W + (size_t)mat * K * Ncols;
    __bf16*      Om = out + (size_t)mat * Ncols * Kp;
    int idx = blockIdx.x * 256 + threadIdx.x;
    if (idx >= Ncols * Kp) return;
    int n = idx / Kp, k = idx % Kp;
    Om[(size_t)n * Kp + k] = (k < K) ? (__bf16)Wm[(size_t)k * Ncols + n] : (__bf16)0.0f;
}

// edge_type = argmax(edge_attr) (first max wins); histogram counts per
// (dst, rel) segment with native f32 atomics.
__global__ void edge_prep_kernel(const float* __restrict__ edge_attr,
                                 const int* __restrict__ edge_index,
                                 int* __restrict__ etype, float* __restrict__ cnt) {
    int e = blockIdx.x * 256 + threadIdx.x;
    if (e >= EE) return;
    float4 a = *(const float4*)(edge_attr + (size_t)e * 4);
    int t = 0; float best = a.x;
    if (a.y > best) { best = a.y; t = 1; }
    if (a.z > best) { best = a.z; t = 2; }
    if (a.w > best) { best = a.w; t = 3; }
    etype[e] = t;
    int dst = edge_index[EE + e];
    unsafeAtomicAdd(cnt + (size_t)dst * RR + t, 1.0f);
}

__global__ void invcnt_kernel(float* cnt) {
    int i = blockIdx.x * 256 + threadIdx.x;
    if (i >= NN * RR) return;
    cnt[i] = 1.0f / fmaxf(cnt[i], 1.0f);
}

// One wave (32 lanes) per edge; lane handles 4 consecutive bf16 features.
// Gathers h[src] (coalesced 256B bf16 row) and f32-scatter-adds into agg.
__global__ void aggregate_kernel(const __bf16* __restrict__ h,
                                 const int* __restrict__ edge_index,
                                 const int* __restrict__ etype,
                                 float* __restrict__ agg) {
    int e = blockIdx.x * 8 + (threadIdx.x >> 5);
    if (e >= EE) return;
    int lane = threadIdx.x & 31;
    int src  = edge_index[e];
    int dst  = edge_index[EE + e];
    int t    = etype[e];
    const v4bf v = *(const v4bf*)(h + (size_t)src * HH + lane * 4);
    float* d = agg + ((size_t)dst * RR + t) * HH + lane * 4;
    unsafeAtomicAdd(d + 0, (float)v[0]);
    unsafeAtomicAdd(d + 1, (float)v[1]);
    unsafeAtomicAdd(d + 2, (float)v[2]);
    unsafeAtomicAdd(d + 3, (float)v[3]);
}

// ---- embedder: h = relu(x@W1 + b1) @ W2 + b2, both GEMMs on WMMA -----------
__global__ void __launch_bounds__(256)
embed_kernel(const float* __restrict__ x,
             const __bf16* __restrict__ W1t,   // [128][32] (K padded 16->32)
             const float* __restrict__ b1,
             const __bf16* __restrict__ W2t,   // [128][128]
             const float* __restrict__ b2,
             __bf16* __restrict__ h_out) {
    __shared__ __bf16 sX[16][40];      // 32 + 8 pad (bank de-conflict)
    __shared__ __bf16 sH1[16][136];    // 128 + 8 pad
    const int n0 = blockIdx.x * 16;
    const int tid = threadIdx.x;

    for (int idx = tid; idx < 16 * 32; idx += 256) {
        int m = idx >> 5, k = idx & 31;
        sX[m][k] = (k < FN) ? (__bf16)x[(size_t)(n0 + m) * FN + k] : (__bf16)0.0f;
    }
    __syncthreads();

    const int wave = tid >> 5, lane = tid & 31;
    const int ncol0 = wave * 16;
    const int n  = ncol0 + (lane & 15);
    const int mb = (lane >> 4) * 8;

    v8f acc = {};
    {
        v16bf a = lds_a_frag(&sX[0][0], 40, 0);
        v16bf b = gbl_b_frag(W1t, 32, ncol0, 0);
        acc = __builtin_amdgcn_wmma_f32_16x16x32_bf16(false, a, false, b,
                                                      (short)0, acc, false, false);
    }
    float bv = b1[n];
#pragma unroll
    for (int g = 0; g < 8; g++)
        sH1[g + mb][n] = (__bf16)fmaxf(acc[g] + bv, 0.0f);
    __syncthreads();

    v8f acc2 = {};
#pragma unroll
    for (int kt = 0; kt < 4; kt++) {
        v16bf a = lds_a_frag(&sH1[0][0], 136, kt);
        v16bf b = gbl_b_frag(W2t, 128, ncol0, kt);
        acc2 = __builtin_amdgcn_wmma_f32_16x16x32_bf16(false, a, false, b,
                                                       (short)0, acc2, false, false);
    }
    float bv2 = b2[n];
#pragma unroll
    for (int g = 0; g < 8; g++)
        h_out[(size_t)(n0 + g + mb) * HH + n] = (__bf16)(acc2[g] + bv2);
}

// ---- fused R-GCN update: h' = sum_r (agg_r * invcnt_r) @ W_r + h @ W_root + b
__global__ void __launch_bounds__(256)
rgcn_update_kernel(const __bf16* __restrict__ h_in,
                   const float* __restrict__ agg,       // [N][R][H] f32
                   const float* __restrict__ invcnt,    // [N][R]
                   const __bf16* __restrict__ relWt,    // [R][H][H] transposed bf16
                   const __bf16* __restrict__ rootWt,   // [H][H]    transposed bf16
                   const float* __restrict__ bias,      // [H]
                   __bf16* __restrict__ h_out,
                   int do_relu) {
    __shared__ __bf16 sA[5][16][136];  // 4 scaled agg tiles + h tile, padded
    const int n0 = blockIdx.x * 16;
    const int tid = threadIdx.x;

    // Hint the prefetcher at this block's weight columns.
    __builtin_prefetch(relWt + (size_t)((tid >> 5) * 16) * HH, 0, 3);

    // --- h tile (bf16, raw byte copy): async DMA straight into LDS.
    // Tile = 16 rows x 256B = 256 chunks of 16B; one chunk per thread.
    // LDS dest addresses honor the 136-element (272B) anti-bank-conflict pad.
    {
        int m  = tid >> 4;         // row 0..15
        int kc = tid & 15;         // 16B chunk within row
        unsigned laddr = (unsigned)(size_t)(void*)&sA[4][m][kc * 8];
        const __bf16* gaddr = h_in + (size_t)(n0 + m) * HH + kc * 8;
        asm volatile("global_load_async_to_lds_b128 %0, %1, off"
                     :: "v"(laddr), "v"(gaddr) : "memory");
    }

    // --- agg tiles: f32 load, scale by 1/cnt, convert bf16 (VALU overlaps DMA)
    for (int idx = tid; idx < 4 * 16 * 32; idx += 256) {   // float4 granules
        int r = idx >> 9, m = (idx >> 5) & 15, kc = idx & 31;
        float ic = invcnt[(size_t)(n0 + m) * RR + r];
        v4f a = *(const v4f*)(agg + ((size_t)(n0 + m) * RR + r) * HH + kc * 4);
        v4bf o;
        o[0] = (__bf16)(a[0] * ic); o[1] = (__bf16)(a[1] * ic);
        o[2] = (__bf16)(a[2] * ic); o[3] = (__bf16)(a[3] * ic);
        *(v4bf*)&sA[r][m][kc * 4] = o;
    }

    // Drain this wave's async DMA, then make all waves' LDS writes visible.
    asm volatile("s_wait_asynccnt 0x0" ::: "memory");
    __syncthreads();

    const int wave = tid >> 5, lane = tid & 31;
    const int ncol0 = wave * 16;

    v8f acc = {};
#pragma unroll
    for (int s = 0; s < 5; s++) {
        const __bf16* Wt = (s < 4) ? (relWt + (size_t)s * HH * HH) : rootWt;
#pragma unroll
        for (int kt = 0; kt < 4; kt++) {
            v16bf a = lds_a_frag(&sA[s][0][0], 136, kt);
            v16bf b = gbl_b_frag(Wt, 128, ncol0, kt);
            acc = __builtin_amdgcn_wmma_f32_16x16x32_bf16(false, a, false, b,
                                                          (short)0, acc, false, false);
        }
    }

    const int n  = ncol0 + (lane & 15);
    const int mb = (lane >> 4) * 8;
    float bv = bias[n];
#pragma unroll
    for (int g = 0; g < 8; g++) {
        float v = acc[g] + bv;
        if (do_relu) v = fmaxf(v, 0.0f);
        h_out[(size_t)(n0 + g + mb) * HH + n] = (__bf16)v;
    }
}

// ---- global add pool + MLP head --------------------------------------------
__global__ void pool_kernel(const __bf16* __restrict__ h,
                            const int* __restrict__ batch_vec,
                            float* __restrict__ pooled) {
    int idx = blockIdx.x * 256 + threadIdx.x;
    if (idx >= NN * HH) return;
    int node = idx >> 7, k = idx & 127;
    unsafeAtomicAdd(pooled + (size_t)batch_vec[node] * HH + k, (float)h[idx]);
}

__global__ void head_kernel(const float* __restrict__ pooled,
                            const float* __restrict__ w1, const float* __restrict__ b1,
                            const float* __restrict__ w2, const float* __restrict__ b2,
                            float* __restrict__ out) {
    __shared__ float sp[HH];
    __shared__ float red[HH];
    const int g = blockIdx.x, t = threadIdx.x;   // 128 threads
    sp[t] = pooled[(size_t)g * HH + t];
    __syncthreads();
    float acc = b1[t];
#pragma unroll 8
    for (int k = 0; k < HH; k++) acc = fmaf(sp[k], w1[(size_t)k * HH + t], acc);
    red[t] = fmaxf(acc, 0.0f) * w2[t];           // head_w2 is [128][1]
    __syncthreads();
    for (int s = 64; s > 0; s >>= 1) {
        if (t < s) red[t] += red[t + s];
        __syncthreads();
    }
    if (t == 0) out[g] = red[0] + b2[0];
}

// ---------------------------------------------------------------------------

extern "C" void kernel_launch(void* const* d_in, const int* in_sizes, int n_in,
                              void* d_out, int out_size, void* d_ws, size_t ws_size,
                              hipStream_t stream) {
    const float* x          = (const float*)d_in[0];
    const int*   edge_index = (const int*)  d_in[1];
    const float* edge_attr  = (const float*)d_in[2];
    const int*   batch_vec  = (const int*)  d_in[3];
    const float* emb_w1     = (const float*)d_in[4];
    const float* emb_b1     = (const float*)d_in[5];
    const float* emb_w2     = (const float*)d_in[6];
    const float* emb_b2     = (const float*)d_in[7];
    const float* rel_w      = (const float*)d_in[8];
    const float* root_w     = (const float*)d_in[9];
    const float* rgcn_b     = (const float*)d_in[10];
    const float* head_w1    = (const float*)d_in[11];
    const float* head_b1    = (const float*)d_in[12];
    const float* head_w2    = (const float*)d_in[13];
    const float* head_b2    = (const float*)d_in[14];
    float* out = (float*)d_out;

    // ---- workspace carve-up (all offsets 256B aligned) ----
    char* base = (char*)d_ws;
    size_t off = 0;
    auto carve = [&](size_t bytes) -> char* {
        char* p = base + off;
        off += (bytes + 255) & ~(size_t)255;
        return p;
    };
    __bf16* hA     = (__bf16*)carve((size_t)NN * HH * 2);
    __bf16* hB     = (__bf16*)carve((size_t)NN * HH * 2);
    float*  agg    = (float*) carve((size_t)NN * RR * HH * 4);
    float*  cnt    = (float*) carve((size_t)NN * RR * 4);          // -> invcnt
    int*    etype  = (int*)   carve((size_t)EE * 4);
    float*  pooled = (float*) carve((size_t)GG * HH * 4);
    __bf16* W1t    = (__bf16*)carve((size_t)HH * 32 * 2);
    __bf16* W2t    = (__bf16*)carve((size_t)HH * HH * 2);
    __bf16* relWt  = (__bf16*)carve((size_t)DEPTH * RR * HH * HH * 2);
    __bf16* rootWt = (__bf16*)carve((size_t)DEPTH * HH * HH * 2);
    (void)ws_size; (void)in_sizes; (void)n_in; (void)out_size;

    // ---- prep: zero scratch, pack weights, edge types + counts ----
    zero4_kernel<<<196, 256, 0, stream>>>(cnt,    (size_t)NN * RR / 4);
    zero4_kernel<<<8,   256, 0, stream>>>(pooled, (size_t)GG * HH / 4);

    pack_bt_kernel<<<dim3((HH * 32 + 255) / 256, 1),  256, 0, stream>>>(emb_w1, W1t,  FN, HH, 32);
    pack_bt_kernel<<<dim3((HH * HH + 255) / 256, 1),  256, 0, stream>>>(emb_w2, W2t,  HH, HH, HH);
    pack_bt_kernel<<<dim3((HH * HH + 255) / 256, DEPTH * RR), 256, 0, stream>>>(rel_w,  relWt,  HH, HH, HH);
    pack_bt_kernel<<<dim3((HH * HH + 255) / 256, DEPTH),      256, 0, stream>>>(root_w, rootWt, HH, HH, HH);

    edge_prep_kernel<<<(EE + 255) / 256, 256, 0, stream>>>(edge_attr, edge_index, etype, cnt);
    invcnt_kernel<<<(NN * RR + 255) / 256, 256, 0, stream>>>(cnt);   // in-place -> 1/max(cnt,1)

    // ---- embedder ----
    embed_kernel<<<NN / 16, 256, 0, stream>>>(x, W1t, emb_b1, W2t, emb_b2, hA);

    // ---- R-GCN layers ----
    __bf16* hcur = hA;
    __bf16* hnxt = hB;
    for (int d = 0; d < DEPTH; d++) {
        zero4_kernel<<<6400, 256, 0, stream>>>(agg, (size_t)NN * RR * HH / 4);
        aggregate_kernel<<<EE / 8, 256, 0, stream>>>(hcur, edge_index, etype, agg);
        rgcn_update_kernel<<<NN / 16, 256, 0, stream>>>(
            hcur, agg, cnt,
            relWt  + (size_t)d * RR * HH * HH,
            rootWt + (size_t)d * HH * HH,
            rgcn_b + (size_t)d * HH,
            hnxt, (d != DEPTH - 1) ? 1 : 0);
        __bf16* t = hcur; hcur = hnxt; hnxt = t;
    }

    // ---- pool + head ----
    pool_kernel<<<(NN * HH + 255) / 256, 256, 0, stream>>>(hcur, batch_vec, pooled);
    head_kernel<<<GG, 128, 0, stream>>>(pooled, head_w1, head_b1, head_w2, head_b2, out);
}
